// FusedSparseMoeBlock_48902497632320
// MI455X (gfx1250) — matrix operand order repaired and checked by reading, not compiled, not run on hardware
//
#include <hip/hip_runtime.h>
#include <hip/hip_bf16.h>
#include <math.h>

typedef __attribute__((ext_vector_type(16))) __bf16 v16bf;
typedef __attribute__((ext_vector_type(8)))  float  v8f;
typedef __attribute__((ext_vector_type(4)))  int    v4i_t;

#define M_TOK 512
#define H_DIM 2048
#define E_NUM 8
#define I_DIM 5632
#define TWOI  11264
#define KC    64    // K-chunk per LDS stage
#define LDSP  72    // halves per LDS row: 64 + 8 pad -> 144B rows (16B aligned, 36-bank stride)

#if __has_builtin(__builtin_amdgcn_global_load_async_to_lds_b128) && __has_builtin(__builtin_amdgcn_s_wait_asynccnt)
#define HAS_ASYNC_LDS 1
typedef __attribute__((address_space(1))) v4i_t* gv4i_p;
typedef __attribute__((address_space(3))) v4i_t* sv4i_p;
#else
#define HAS_ASYNC_LDS 0
#endif

__device__ __forceinline__ unsigned short f2bf(float f) {
  union { float f; unsigned u; } c; c.f = f;
  unsigned u = c.u;
  u = (u + 0x7FFFu + ((u >> 16) & 1u)) >> 16;   // round-to-nearest-even
  return (unsigned short)u;
}

__device__ __forceinline__ unsigned pack2bf(float lo, float hi) {
  return (unsigned)f2bf(lo) | ((unsigned)f2bf(hi) << 16);
}

__device__ __forceinline__ v16bf load_frag(const unsigned short* p0, const unsigned short* p1) {
  union { uint4 q[2]; v16bf v; } fc;
  fc.q[0] = *(const uint4*)p0;
  fc.q[1] = *(const uint4*)p1;
  return fc.v;
}

__device__ __forceinline__ float fast_silu(float g) {
  // g * sigmoid(g) with v_rcp_f32 (no IEEE div fix-up chain)
  return g * __builtin_amdgcn_rcpf(1.f + __expf(-g));
}

// ---------------- init ----------------
__global__ void moe_init(int* cnt) {
  if (threadIdx.x < E_NUM) cnt[threadIdx.x] = 0;
}

// ---------------- router: logits -> softmax -> top2 -> renorm -> compaction ----------------
__global__ __launch_bounds__(256) void moe_router(
    const float* __restrict__ x, const float* __restrict__ wg,
    float* __restrict__ combine, int* __restrict__ cnt,
    int* __restrict__ tok_ids, int* __restrict__ slot_of) {
  const int wave = threadIdx.x >> 5, lane = threadIdx.x & 31;
  const int m = blockIdx.x * 8 + wave;
  const float* xr = x + (size_t)m * H_DIM;

  float acc[E_NUM];
#pragma unroll
  for (int e = 0; e < E_NUM; ++e) acc[e] = 0.f;
  for (int h = lane; h < H_DIM; h += 32) {
    float xv = xr[h];
    const float* wr = wg + (size_t)h * E_NUM;
#pragma unroll
    for (int e = 0; e < E_NUM; ++e) acc[e] += xv * wr[e];
  }
#pragma unroll
  for (int e = 0; e < E_NUM; ++e)
#pragma unroll
    for (int off = 16; off > 0; off >>= 1)
      acc[e] += __shfl_xor(acc[e], off, 32);

  if (lane == 0) {
    float mx = acc[0];
#pragma unroll
    for (int e = 1; e < E_NUM; ++e) mx = fmaxf(mx, acc[e]);
    float p[E_NUM], s = 0.f;
#pragma unroll
    for (int e = 0; e < E_NUM; ++e) { p[e] = __expf(acc[e] - mx); s += p[e]; }
    float inv = 1.f / s;
#pragma unroll
    for (int e = 0; e < E_NUM; ++e) p[e] *= inv;
    int e0 = 0;
#pragma unroll
    for (int e = 1; e < E_NUM; ++e) if (p[e] > p[e0]) e0 = e;
    int e1 = (e0 == 0) ? 1 : 0;
#pragma unroll
    for (int e = 0; e < E_NUM; ++e) if (e != e0 && p[e] > p[e1]) e1 = e;
    float rs = 1.f / (p[e0] + p[e1]);
    float w0 = p[e0] * rs, w1 = p[e1] * rs;
#pragma unroll
    for (int e = 0; e < E_NUM; ++e)
      combine[m * E_NUM + e] = (e == e0) ? w0 : ((e == e1) ? w1 : 0.f);
    int i0 = atomicAdd(&cnt[e0], 1);
    tok_ids[e0 * M_TOK + i0] = m;
    slot_of[2 * m] = e0 * M_TOK + i0;
    int i1 = atomicAdd(&cnt[e1], 1);
    tok_ids[e1 * M_TOK + i1] = m;
    slot_of[2 * m + 1] = e1 * M_TOK + i1;
  }
}

// ---------------- grouped GEMM1 (gate_up) + silu*mul, bf16 WMMA ----------------
// grid: (I/256, M/16, E). block 256 = 8 waves; each wave: 16x32 gate + 16x32 up; K staged 64-deep.
__global__ __launch_bounds__(256) void moe_gemm1_silu(
    const float* __restrict__ x, const float* __restrict__ w1,
    const int* __restrict__ cnt, const int* __restrict__ tok_ids,
    unsigned short* __restrict__ hbuf) {
  __shared__ __align__(16) unsigned short As[16 * LDSP];
  __shared__ __align__(16) unsigned short Bg[256 * LDSP];
  __shared__ __align__(16) unsigned short Bu[256 * LDSP];
  __shared__ int toks[16];

  const int e = blockIdx.z, mt = blockIdx.y;
  const int rows = cnt[e];
  if (mt * 16 >= rows) return;
  const int tid = threadIdx.x, lane = tid & 31, wave = tid >> 5;
  if (tid < 16) {
    int r = mt * 16 + tid;
    toks[tid] = tok_ids[e * M_TOK + (r < rows ? r : 0)];
  }
  __syncthreads();

  const int cg0 = blockIdx.x * 256;
  const float* w1e = w1 + (size_t)e * H_DIM * TWOI;

  v8f acc_g0 = {}, acc_g1 = {}, acc_u0 = {}, acc_u1 = {};
  const int r16   = lane & 15;
  const int abase = (lane < 16) ? 0 : 16;  // A-frag K sub-offset (contiguous k = 16h + s, same as B)
  const int kbase = (lane < 16) ? 0 : 16;  // B-frag K base (contiguous k = 16h + s, same as A)
  const int c     = wave * 32 + r16;

  for (int k0 = 0; k0 < H_DIM; k0 += KC) {
    // stage A: 16 gathered token rows x 64 K (fp32 -> bf16, packed b32 stores)
    for (int idx = tid; idx < 16 * (KC / 2); idx += 256) {
      int r = idx >> 5, kp = idx & 31;            // 32 pairs per row
      const float* xp = x + (size_t)toks[r] * H_DIM + k0 + 2 * kp;
      ((unsigned*)&As[r * LDSP])[kp] = pack2bf(xp[0], xp[1]);
    }
    // stage B (gate & up), transposed [col][k]
    {
      const float* bg = w1e + (size_t)k0 * TWOI + cg0 + tid;
      const float* bu = bg + I_DIM;
      unsigned* BgRow = (unsigned*)&Bg[tid * LDSP];
      unsigned* BuRow = (unsigned*)&Bu[tid * LDSP];
#pragma unroll 4
      for (int k = 0; k < KC; k += 2) {
        float g0v = bg[(size_t)k * TWOI], g1v = bg[(size_t)(k + 1) * TWOI];
        float u0v = bu[(size_t)k * TWOI], u1v = bu[(size_t)(k + 1) * TWOI];
        BgRow[k >> 1] = pack2bf(g0v, g1v);
        BuRow[k >> 1] = pack2bf(u0v, u1v);
        __builtin_prefetch(bg + (size_t)(k + KC) * TWOI, 0, 0);   // next K-chunk
      }
    }
    __syncthreads();

#pragma unroll
    for (int kh = 0; kh < 2; ++kh) {
      const int ka = kh * 32;
      v16bf a   = load_frag(&As[r16 * LDSP + ka + abase], &As[r16 * LDSP + ka + abase + 8]);
      v16bf bg0 = load_frag(&Bg[c * LDSP + ka + kbase], &Bg[c * LDSP + ka + kbase + 8]);
      v16bf bg1 = load_frag(&Bg[(c + 16) * LDSP + ka + kbase], &Bg[(c + 16) * LDSP + ka + kbase + 8]);
      v16bf bu0 = load_frag(&Bu[c * LDSP + ka + kbase], &Bu[c * LDSP + ka + kbase + 8]);
      v16bf bu1 = load_frag(&Bu[(c + 16) * LDSP + ka + kbase], &Bu[(c + 16) * LDSP + ka + kbase + 8]);
      acc_g0 = __builtin_amdgcn_wmma_f32_16x16x32_bf16(false, a, false, bg0, (short)0, acc_g0, false, false);
      acc_g1 = __builtin_amdgcn_wmma_f32_16x16x32_bf16(false, a, false, bg1, (short)0, acc_g1, false, false);
      acc_u0 = __builtin_amdgcn_wmma_f32_16x16x32_bf16(false, a, false, bu0, (short)0, acc_u0, false, false);
      acc_u1 = __builtin_amdgcn_wmma_f32_16x16x32_bf16(false, a, false, bu1, (short)0, acc_u1, false, false);
    }
    __syncthreads();
  }

  // h = silu(g) * u, store bf16
  const int colA = cg0 + wave * 32 + r16;
#pragma unroll
  for (int v = 0; v < 8; ++v) {
    int row = (lane < 16) ? v : (v + 8);      // C/D layout: VGPR v -> M=v / M=v+8
    size_t slot = (size_t)e * M_TOK + mt * 16 + row;
    float h0 = fast_silu(acc_g0[v]) * acc_u0[v];
    float h1 = fast_silu(acc_g1[v]) * acc_u1[v];
    hbuf[slot * I_DIM + colA]      = f2bf(h0);
    hbuf[slot * I_DIM + colA + 16] = f2bf(h1);
  }
}

// ---------------- grouped GEMM2 (down proj), bf16 WMMA, fp32 out per slot ----------------
// grid: (H/256, M/16, E). A tile (already bf16) staged via gfx1250 async global->LDS when available.
__global__ __launch_bounds__(256) void moe_gemm2(
    const unsigned short* __restrict__ hbuf, const float* __restrict__ w2,
    const int* __restrict__ cnt, float* __restrict__ down) {
  __shared__ __align__(16) unsigned short As[16 * LDSP];
  __shared__ __align__(16) unsigned short Bs[256 * LDSP];

  const int e = blockIdx.z, mt = blockIdx.y;
  const int rows = cnt[e];
  if (mt * 16 >= rows) return;
  const int tid = threadIdx.x, lane = tid & 31, wave = tid >> 5;
  const int c0 = blockIdx.x * 256;
  const float* w2e = w2 + (size_t)e * I_DIM * H_DIM;
  const size_t slotbase = (size_t)e * M_TOK + mt * 16;

  v8f acc0 = {}, acc1 = {};
  const int r16   = lane & 15;
  const int abase = (lane < 16) ? 0 : 16;
  const int kbase = (lane < 16) ? 0 : 16;
  const int c     = wave * 32 + r16;

  for (int k0 = 0; k0 < I_DIM; k0 += KC) {
    // stage A: 16 rows x 64 bf16 = 128 x b128 chunks, threads 0..127 (waves 0-3)
    if (tid < 128) {
      int r = tid >> 3, p = tid & 7;
      const unsigned short* src = hbuf + (slotbase + r) * I_DIM + k0 + p * 8;
#if HAS_ASYNC_LDS
      __builtin_amdgcn_global_load_async_to_lds_b128(
          (gv4i_p)src, (sv4i_p)&As[r * LDSP + p * 8], 0, 0);
#else
      *(uint4*)&As[r * LDSP + p * 8] = *(const uint4*)src;
#endif
    }
    // stage B: w2 rows fp32 -> bf16, transposed [col][k]
    {
      const float* bp = w2e + (size_t)k0 * H_DIM + c0 + tid;
      unsigned* BsRow = (unsigned*)&Bs[tid * LDSP];
#pragma unroll 4
      for (int k = 0; k < KC; k += 2) {
        float v0 = bp[(size_t)k * H_DIM], v1 = bp[(size_t)(k + 1) * H_DIM];
        BsRow[k >> 1] = pack2bf(v0, v1);
        __builtin_prefetch(bp + (size_t)(k + KC) * H_DIM, 0, 0);
      }
    }
#if HAS_ASYNC_LDS
    if (wave < 4) __builtin_amdgcn_s_wait_asynccnt(0);
#endif
    __syncthreads();

#pragma unroll
    for (int kh = 0; kh < 2; ++kh) {
      const int ka = kh * 32;
      v16bf a  = load_frag(&As[r16 * LDSP + ka + abase], &As[r16 * LDSP + ka + abase + 8]);
      v16bf b0 = load_frag(&Bs[c * LDSP + ka + kbase], &Bs[c * LDSP + ka + kbase + 8]);
      v16bf b1 = load_frag(&Bs[(c + 16) * LDSP + ka + kbase], &Bs[(c + 16) * LDSP + ka + kbase + 8]);
      acc0 = __builtin_amdgcn_wmma_f32_16x16x32_bf16(false, a, false, b0, (short)0, acc0, false, false);
      acc1 = __builtin_amdgcn_wmma_f32_16x16x32_bf16(false, a, false, b1, (short)0, acc1, false, false);
    }
    __syncthreads();
  }
  const int colc = c0 + wave * 32 + r16;
#pragma unroll
  for (int v = 0; v < 8; ++v) {
    int row = (lane < 16) ? v : (v + 8);
    down[(slotbase + row) * H_DIM + colc]      = acc0[v];
    down[(slotbase + row) * H_DIM + colc + 16] = acc1[v];
  }
}

// ---------------- final deterministic combine: out = w0*down[slot0] + w1*down[slot1] ----------------
__global__ __launch_bounds__(256) void moe_combine(
    const float* __restrict__ down, const float* __restrict__ combine,
    const int* __restrict__ slot_of, float* __restrict__ out) {
  int idx = blockIdx.x * blockDim.x + threadIdx.x;   // over M * (H/4)
  int m = idx >> 9;                                  // H/4 = 512
  int c4 = idx & 511;
  int s0 = slot_of[2 * m], s1 = slot_of[2 * m + 1];
  float w0 = combine[m * E_NUM + (s0 >> 9)];         // slot = e*512 + i
  float w1 = combine[m * E_NUM + (s1 >> 9)];
  float4 d0 = ((const float4*)(down + (size_t)s0 * H_DIM))[c4];
  float4 d1 = ((const float4*)(down + (size_t)s1 * H_DIM))[c4];
  float4 o;
  o.x = w0 * d0.x + w1 * d1.x;
  o.y = w0 * d0.y + w1 * d1.y;
  o.z = w0 * d0.z + w1 * d1.z;
  o.w = w0 * d0.w + w1 * d1.w;
  ((float4*)out)[idx] = o;
}

extern "C" void kernel_launch(void* const* d_in, const int* in_sizes, int n_in,
                              void* d_out, int out_size, void* d_ws, size_t ws_size,
                              hipStream_t stream) {
  const float* x   = (const float*)d_in[0];  // [1,512,2048]
  const float* wg  = (const float*)d_in[1];  // [2048,8]
  const float* w1p = (const float*)d_in[2];  // [8,2048,11264]
  const float* w2p = (const float*)d_in[3];  // [8,5632,2048]
  float* out = (float*)d_out;
  (void)in_sizes; (void)n_in; (void)out_size; (void)ws_size;

  // workspace layout (256B aligned sections)
  char* ws = (char*)d_ws;
  int*   cnt     = (int*)(ws + 0);                         // 8 ints
  float* combine = (float*)(ws + 256);                     // 512*8 f32   = 16384B
  int*   tok_ids = (int*)(ws + 16640);                     // 8*512 ints  = 16384B
  int*   slot_of = (int*)(ws + 33024);                     // 512*2 ints  =  4096B
  unsigned short* hbuf = (unsigned short*)(ws + 37120);    // 8*512*5632 bf16 = 46,137,344B
  float* down    = (float*)(ws + 37120 + 46137344ull);     // 8*512*2048 f32  = 33,554,432B

  moe_init<<<1, 32, 0, stream>>>(cnt);
  moe_router<<<64, 256, 0, stream>>>(x, wg, combine, cnt, tok_ids, slot_of);
  moe_gemm1_silu<<<dim3(I_DIM / 256, M_TOK / 16, E_NUM), 256, 0, stream>>>(x, w1p, cnt, tok_ids, hbuf);
  moe_gemm2<<<dim3(H_DIM / 256, M_TOK / 16, E_NUM), 256, 0, stream>>>(hbuf, w2p, cnt, down);
  moe_combine<<<(M_TOK * (H_DIM / 4)) / 256, 256, 0, stream>>>(down, combine, slot_of, out);
}
